// Block_88244398064014
// MI455X (gfx1250) — matrix operand order repaired and checked by reading, not compiled, not run on hardware
//
#include <hip/hip_runtime.h>
#include <hip/hip_bf16.h>

#define D_   1024
#define H_   16
#define HD_  64
#define DFF_ 4096
#define B_   2
#define S_   2048
#define BS_  (B_*S_)

typedef __bf16 bf16;
typedef __attribute__((ext_vector_type(16))) __bf16 v16bf;
typedef __attribute__((ext_vector_type(8)))  __bf16 v8bf;
typedef __attribute__((ext_vector_type(8)))  float  v8f;
typedef unsigned int u32x4 __attribute__((ext_vector_type(4)));
typedef int          i32x4 __attribute__((ext_vector_type(4)));
typedef int          i32x8 __attribute__((ext_vector_type(8)));

static __device__ inline v8f wmma_bf16(v16bf a, v16bf b, v8f c) {
  // D = A(16x32 bf16) * B(32x16 bf16) + C(16x16 f32)
  return __builtin_amdgcn_wmma_f32_16x16x32_bf16(false, a, false, b, (short)0, c,
                                                 false, false);
}

// A-fragment of a row-major MxK matrix (16 rows x 32 K), CDNA5 16-bit layout:
//   lanes 0-15 : row=lane,     K = kbase+{0..7, 16..23}
//   lanes 16-31: row=lane-16,  K = kbase+{8..15, 24..31}
static __device__ inline v16bf load_fragA(const bf16* base, int stride,
                                          int rowTile, int kbase, int lane) {
  int r  = rowTile + (lane & 15);
  int k0 = kbase + ((lane & 16) ? 8 : 0);
  const bf16* p = base + (size_t)r * stride + k0;
  v8bf lo = *(const v8bf*)(p);
  v8bf hi = *(const v8bf*)(p + 16);
  v16bf out;
#pragma unroll
  for (int i = 0; i < 8; ++i) { out[i] = lo[i]; out[i + 8] = hi[i]; }
  return out;
}

// B-fragment read from Bt = B^T stored row-major (N x K):
//   lanes 0-15 : col=lane,     K = kbase+{0..7, 16..23}
//   lanes 16-31: col=lane-16,  K = kbase+{8..15, 24..31}
static __device__ inline v16bf load_fragB(const bf16* Bt, int stride,
                                          int nTile, int kbase, int lane) {
  int n  = nTile + (lane & 15);
  int k0 = kbase + ((lane & 16) ? 8 : 0);
  const bf16* p = Bt + (size_t)n * stride + k0;
  v8bf lo = *(const v8bf*)(p);
  v8bf hi = *(const v8bf*)(p + 16);
  v16bf out;
#pragma unroll
  for (int i = 0; i < 8; ++i) { out[i] = lo[i]; out[i + 8] = hi[i]; }
  return out;
}

// ---- Tensor Data Mover: DMA a 2-D tile (tile_d1 rows x tile_d0 elems of 2B)
// from a row-major tensor (row stride = stride0 elems) into LDS. ISA 8.3/8.4.
static __device__ inline void tdm_load_tile_2d(unsigned int lds_byte_addr,
                                               const void* gsrc,
                                               int tensor_d0, int tensor_d1,
                                               int tile_d0, int tile_d1,
                                               int stride0) {
  unsigned long long ga = (unsigned long long)(size_t)gsrc;
  u32x4 g0;
  g0[0] = 1u;                                              // count=1, user D#
  g0[1] = lds_byte_addr;                                   // lds_addr
  g0[2] = (unsigned int)(ga & 0xFFFFFFFFull);              // global_addr lo
  g0[3] = (unsigned int)((ga >> 32) & 0x1FFFFFFull) | (2u << 30); // hi | type=2
  i32x8 g1;
  g1[0] = (int)(1u << 16);                                 // data_size=1 (2B)
  g1[1] = (int)(((unsigned)tensor_d0 & 0xFFFFu) << 16);    // tensor_dim0 lo
  g1[2] = (int)((((unsigned)tensor_d0 >> 16) & 0xFFFFu) |
                (((unsigned)tensor_d1 & 0xFFFFu) << 16));  // dim0 hi | dim1 lo
  g1[3] = (int)((((unsigned)tensor_d1 >> 16) & 0xFFFFu) |
                ((unsigned)tile_d0 << 16));                // dim1 hi | tile_dim0
  g1[4] = (int)((unsigned)tile_d1 & 0xFFFFu);              // tile_dim1, tile_dim2=0
  g1[5] = (int)stride0;                                    // tensor_dim0_stride lo
  g1[6] = 0;                                               // stride0 hi | dim1_stride
  g1[7] = 0;
  i32x4 z4 = {0, 0, 0, 0};
#if defined(__clang_major__) && (__clang_major__ >= 23)
  i32x8 z8 = {0, 0, 0, 0, 0, 0, 0, 0};
  __builtin_amdgcn_tensor_load_to_lds(g0, g1, z4, z4, z8, 0);
#else
  __builtin_amdgcn_tensor_load_to_lds(g0, g1, z4, z4, 0);
#endif
}

// ---- fp32 [batch][K][N] -> bf16 [batch][N][K] ------------------------------
__global__ __launch_bounds__(256)
void transpose_cast(const float* __restrict__ src, bf16* __restrict__ dst,
                    int K, int N) {
  size_t base = (size_t)blockIdx.y * K * N;
  int idx = blockIdx.x * 256 + threadIdx.x;
  if (idx < K * N) {
    int k = idx / N, n = idx % N;
    dst[base + (size_t)n * K + k] = (bf16)src[base + idx];
  }
}

// ---- layernorm: fp32 row -> bf16 row ---------------------------------------
__global__ __launch_bounds__(256)
void ln_kernel(const float* __restrict__ x, const float* __restrict__ g,
               const float* __restrict__ beta, bf16* __restrict__ out) {
  int row = blockIdx.x;
  const float* xr = x + (size_t)row * D_;
  __shared__ float red[2][8];
  float s = 0.f, ss = 0.f;
  for (int i = threadIdx.x; i < D_; i += 256) { float v = xr[i]; s += v; ss += v * v; }
#pragma unroll
  for (int off = 16; off; off >>= 1) {
    s  += __shfl_xor(s,  off, 32);
    ss += __shfl_xor(ss, off, 32);
  }
  int wave = threadIdx.x >> 5, lane = threadIdx.x & 31;
  if (lane == 0) { red[0][wave] = s; red[1][wave] = ss; }
  __syncthreads();
  s = 0.f; ss = 0.f;
#pragma unroll
  for (int w = 0; w < 8; ++w) { s += red[0][w]; ss += red[1][w]; }
  float mu  = s / (float)D_;
  float var = ss / (float)D_ - mu * mu;
  float inv = rsqrtf(var + 1e-5f);
  for (int i = threadIdx.x; i < D_; i += 256)
    out[(size_t)row * D_ + i] = (bf16)((xr[i] - mu) * inv * g[i] + beta[i]);
}

// ---- QKV projection: per-head GEMM, V stored transposed --------------------
__global__ __launch_bounds__(128)
void qkv_kernel(const bf16* __restrict__ hb, const bf16* __restrict__ WT,
                bf16* __restrict__ q, bf16* __restrict__ k, bf16* __restrict__ vT) {
  int which = blockIdx.z;                      // 0=q 1=k 2=v
  int b = blockIdx.x / H_, h = blockIdx.x % H_;
  int lane = threadIdx.x & 31, wave = threadIdx.x >> 5;
  int s0 = (blockIdx.y * 4 + wave) * 16;
  const bf16* Arow = hb + (size_t)b * S_ * D_;                 // [S][D]
  const bf16* Bt   = WT + ((size_t)which * H_ + h) * HD_ * D_; // [HD][D]
  v8f acc[4] = {};
  for (int kk = 0; kk < D_; kk += 32) {
    v16bf a = load_fragA(Arow, D_, s0, kk, lane);
#pragma unroll
    for (int t = 0; t < 4; ++t)
      acc[t] = wmma_bf16(a, load_fragB(Bt, D_, t * 16, kk, lane), acc[t]);
  }
  int n = lane & 15;
  int ro = (lane & 16) ? 8 : 0;
  size_t bh = (size_t)b * H_ + h;
#pragma unroll
  for (int t = 0; t < 4; ++t)
#pragma unroll
    for (int r = 0; r < 8; ++r) {
      int s = s0 + ro + r, hd = t * 16 + n;
      bf16 v = (bf16)acc[t][r];
      if (which == 0)      q[(bh * S_ + s) * HD_ + hd] = v;
      else if (which == 1) k[(bh * S_ + s) * HD_ + hd] = v;
      else                 vT[(bh * HD_ + hd) * S_ + s] = v;
    }
}

// ---- causal flash attention: one wave = 16 query rows ----------------------
__global__ __launch_bounds__(128)
void attn_kernel(const bf16* __restrict__ q, const bf16* __restrict__ k,
                 const bf16* __restrict__ vT, bf16* __restrict__ o) {
  __shared__ __align__(16) bf16 pbuf[4][16 * 32];
  int b = blockIdx.x / H_, h = blockIdx.x % H_;
  int lane = threadIdx.x & 31, wave = threadIdx.x >> 5;
  int s0 = (blockIdx.y * 4 + wave) * 16;
  size_t bh = (size_t)b * H_ + h;
  const bf16* Q  = q  + bh * S_ * HD_;
  const bf16* Km = k  + bh * S_ * HD_;
  const bf16* Vt = vT + bh * HD_ * S_;
  v16bf qf0 = load_fragA(Q, HD_, s0, 0,  lane);
  v16bf qf1 = load_fragA(Q, HD_, s0, 32, lane);
  v8f oacc[4] = {};
  float mrow[8], lrow[8];
#pragma unroll
  for (int r = 0; r < 8; ++r) { mrow[r] = -1e30f; lrow[r] = 0.f; }
  int n = lane & 15;
  int ro = (lane & 16) ? 8 : 0;
  bf16* pl = &pbuf[wave][0];

  for (int j = 0; j < s0 + 16; j += 32) {            // causal key blocks of 32
    v8f st[2] = {};
#pragma unroll
    for (int u = 0; u < 2; ++u) {
      st[u] = wmma_bf16(qf0, load_fragB(Km, HD_, j + u * 16, 0,  lane), st[u]);
      st[u] = wmma_bf16(qf1, load_fragB(Km, HD_, j + u * 16, 32, lane), st[u]);
    }
    float sc[2][8];
#pragma unroll
    for (int u = 0; u < 2; ++u)
#pragma unroll
      for (int r = 0; r < 8; ++r) {
        float v = st[u][r] * 0.125f;                 // 1/sqrt(64)
        int key = j + u * 16 + n;
        sc[u][r] = (key > s0 + ro + r) ? -1e30f : v; // causal mask
      }
#pragma unroll
    for (int r = 0; r < 8; ++r) {
      float tm = fmaxf(sc[0][r], sc[1][r]);
#pragma unroll
      for (int off = 1; off < 16; off <<= 1)
        tm = fmaxf(tm, __shfl_xor(tm, off, 32));
      float mnew  = fmaxf(mrow[r], tm);
      float alpha = __expf(mrow[r] - mnew);
      float p0 = __expf(sc[0][r] - mnew);
      float p1 = __expf(sc[1][r] - mnew);
      sc[0][r] = p0; sc[1][r] = p1;
      float ps = p0 + p1;
#pragma unroll
      for (int off = 1; off < 16; off <<= 1)
        ps += __shfl_xor(ps, off, 32);
      lrow[r] = lrow[r] * alpha + ps;
      mrow[r] = mnew;
#pragma unroll
      for (int t = 0; t < 4; ++t) oacc[t][r] *= alpha;
    }
    // C-layout -> A-fragment layout via per-wave LDS tile (16x32 bf16)
#pragma unroll
    for (int u = 0; u < 2; ++u)
#pragma unroll
      for (int r = 0; r < 8; ++r)
        pl[(ro + r) * 32 + u * 16 + n] = (bf16)sc[u][r];
    asm volatile("s_wait_dscnt 0" ::: "memory");     // LDS stores visible
    v16bf pf = load_fragA(pl, 32, 0, 0, lane);
#pragma unroll
    for (int t = 0; t < 4; ++t)
      oacc[t] = wmma_bf16(pf, load_fragB(Vt, S_, t * 16, j, lane), oacc[t]);
    asm volatile("s_wait_dscnt 0" ::: "memory");     // pf consumed before rewrite
  }
#pragma unroll
  for (int t = 0; t < 4; ++t)
#pragma unroll
    for (int r = 0; r < 8; ++r) {
      int s = s0 + ro + r;
      o[((size_t)b * S_ + s) * D_ + h * HD_ + t * 16 + n] =
          (bf16)(oacc[t][r] / lrow[r]);
    }
}

// ---- generic WMMA GEMM with TDM-staged B tile ------------------------------
// 256 threads = 8 waves; block tile = 128 rows x 64 cols.
// Per k-step: wave 0 DMAs the 64x32 bf16 B^T tile into LDS (tensor_load_to_lds,
// s_wait_tensorcnt), all 8 waves consume it via ds_load fragments.
template <bool RELU, bool RESID, bool OUTF, bool OUTB>
__global__ __launch_bounds__(256)
void gemm_kernel(const bf16* __restrict__ A, const bf16* __restrict__ Bt,
                 const float* __restrict__ bias, const float* __restrict__ resid,
                 float* __restrict__ Cf, bf16* __restrict__ Cb,
                 int M, int N, int K) {
  __shared__ __align__(16) bf16 btile[64 * 32];      // 4 KB staging tile
  int lane = threadIdx.x & 31, wave = threadIdx.x >> 5;
  int n0 = blockIdx.x * 64;
  int m0 = (blockIdx.y * 8 + wave) * 16;
  unsigned int ldsa = (unsigned int)(size_t)&btile[0]; // LDS byte offset
  v8f acc[4] = {};
  for (int kk = 0; kk < K; kk += 32) {
    __syncthreads();                                  // btile free to overwrite
    if (wave == 0) {
      tdm_load_tile_2d(ldsa, Bt + (size_t)n0 * K + kk, K, 64, 32, 64, K);
      __builtin_amdgcn_s_wait_tensorcnt(0);           // DMA complete
    }
    __syncthreads();                                  // btile visible to all
    if (kk + 32 < K)                                  // prefetch next A slice
      __builtin_prefetch(A + (size_t)m0 * K + kk + 32, 0, 0);
    v16bf a = load_fragA(A, K, m0, kk, lane);
#pragma unroll
    for (int t = 0; t < 4; ++t)
      acc[t] = wmma_bf16(a, load_fragB(btile, 32, t * 16, 0, lane), acc[t]);
  }
  int n = lane & 15;
  int ro = (lane & 16) ? 8 : 0;
#pragma unroll
  for (int t = 0; t < 4; ++t) {
    int col = n0 + t * 16 + n;
    float bv = bias[col];
#pragma unroll
    for (int r = 0; r < 8; ++r) {
      size_t idx = (size_t)(m0 + ro + r) * N + col;
      float v = acc[t][r] + bv;
      if (RELU)  v = fmaxf(v, 0.f);
      if (RESID) v += resid[idx];
      if (OUTF)  Cf[idx] = v;
      if (OUTB)  Cb[idx] = (bf16)v;
    }
  }
}

extern "C" void kernel_launch(void* const* d_in, const int* in_sizes, int n_in,
                              void* d_out, int out_size, void* d_ws, size_t ws_size,
                              hipStream_t stream) {
  const float* x     = (const float*)d_in[0];
  const float* Wq    = (const float*)d_in[1];
  const float* Wk    = (const float*)d_in[2];
  const float* Wv    = (const float*)d_in[3];
  const float* Wp    = (const float*)d_in[4];
  const float* bp    = (const float*)d_in[5];
  const float* W1    = (const float*)d_in[6];
  const float* b1    = (const float*)d_in[7];
  const float* W2    = (const float*)d_in[8];
  const float* b2    = (const float*)d_in[9];
  const float* g1    = (const float*)d_in[10];
  const float* be1   = (const float*)d_in[11];
  const float* g2    = (const float*)d_in[12];
  const float* be2   = (const float*)d_in[13];
  float* out = (float*)d_out;

  // workspace layout (bytes), total ~120 MB
  char* w = (char*)d_ws;
  const size_t MB = 1u << 20;
  bf16*  hb     = (bf16*)(w + 0);          //  8 MB  ln1(x) bf16        [BS][D]
  bf16*  qb     = (bf16*)(w + 8 * MB);     //  8 MB  q  [B][H][S][HD]
  bf16*  kb     = (bf16*)(w + 16 * MB);    //  8 MB  k  [B][H][S][HD]
  bf16*  vtb    = (bf16*)(w + 24 * MB);    //  8 MB  v^T[B][H][HD][S]
  bf16*  ob     = (bf16*)(w + 32 * MB);    //  8 MB  attn out bf16      [BS][D]
  float* x1     = (float*)(w + 40 * MB);   // 16 MB  x + proj (fp32)    [BS][D]
  bf16*  h2b    = (bf16*)(w + 56 * MB);    //  8 MB  ln2(x1) bf16
  bf16*  actb   = (bf16*)(w + 64 * MB);    // 32 MB  relu(ffn1) bf16    [BS][DFF]
  bf16*  WqkvT  = (bf16*)(w + 96 * MB);    //  6 MB  [3][H][HD][D]
  bf16*  WpT    = (bf16*)(w + 102 * MB);   //  2 MB  [D][D] (N x K)
  bf16*  W1T    = (bf16*)(w + 104 * MB);   //  8 MB  [DFF][D]
  bf16*  W2T    = (bf16*)(w + 112 * MB);   //  8 MB  [D][DFF]
  (void)in_sizes; (void)n_in; (void)out_size; (void)ws_size;

  // 1) weight transpose+cast to bf16
  transpose_cast<<<dim3(D_ * HD_ / 256, H_), 256, 0, stream>>>(Wq, WqkvT,                     D_, HD_);
  transpose_cast<<<dim3(D_ * HD_ / 256, H_), 256, 0, stream>>>(Wk, WqkvT + 1 * H_ * HD_ * D_, D_, HD_);
  transpose_cast<<<dim3(D_ * HD_ / 256, H_), 256, 0, stream>>>(Wv, WqkvT + 2 * H_ * HD_ * D_, D_, HD_);
  transpose_cast<<<dim3(D_ * D_   / 256, 1), 256, 0, stream>>>(Wp, WpT, D_, D_);
  transpose_cast<<<dim3(D_ * DFF_ / 256, 1), 256, 0, stream>>>(W1, W1T, D_, DFF_);
  transpose_cast<<<dim3(D_ * DFF_ / 256, 1), 256, 0, stream>>>(W2, W2T, DFF_, D_);

  // 2) pre-LN 1
  ln_kernel<<<BS_, 256, 0, stream>>>(x, g1, be1, hb);

  // 3) QKV projections (WMMA)
  qkv_kernel<<<dim3(B_ * H_, S_ / 64, 3), 128, 0, stream>>>(hb, WqkvT, qb, kb, vtb);

  // 4) causal flash attention (WMMA)
  attn_kernel<<<dim3(B_ * H_, S_ / 64), 128, 0, stream>>>(qb, kb, vtb, ob);

  // 5) output projection + residual -> x1 (fp32)
  gemm_kernel<false, true, true, false>
      <<<dim3(D_ / 64, BS_ / 128), 256, 0, stream>>>(ob, WpT, bp, x, x1, nullptr,
                                                     BS_, D_, D_);
  // 6) pre-LN 2
  ln_kernel<<<BS_, 256, 0, stream>>>(x1, g2, be2, h2b);

  // 7) FFN1 + ReLU -> bf16 activations
  gemm_kernel<true, false, false, true>
      <<<dim3(DFF_ / 64, BS_ / 128), 256, 0, stream>>>(h2b, W1T, b1, nullptr, nullptr,
                                                       actb, BS_, DFF_, D_);
  // 8) FFN2 + residual -> d_out (fp32)
  gemm_kernel<false, true, true, false>
      <<<dim3(D_ / 64, BS_ / 128), 256, 0, stream>>>(actb, W2T, b2, x1, out, nullptr,
                                                     BS_, D_, DFF_);
}
